// Field_64682207478165
// MI455X (gfx1250) — matrix operand order repaired and checked by reading, not compiled, not run on hardware
//
#include <hip/hip_runtime.h>
#include <hip/hip_bf16.h>

// ---------------------------------------------------------------------------
// Instant-NGP style Field forward for gfx1250 (MI455X, wave32, WMMA).
//   - hash-grid encode: 16 levels, F=2, T=2^19, trilinear (8 corners)
//   - MLP: (35 -> 32) ELU, (32 -> 32) ELU, (32 -> 1)
// One wave = one 16-point tile. MLP layers 0/1 run on v_wmma_f32_16x16x32_f16.
// The full hash table (64 MB) is L2-resident on MI455X (192 MB L2); streaming
// x/out traffic is issued with non-temporal hints so it does not evict it.
// ---------------------------------------------------------------------------

typedef _Float16 v16h __attribute__((ext_vector_type(16)));
typedef float    v8f  __attribute__((ext_vector_type(8)));

union Frag16 { unsigned int u[8]; v16h h; };

#define NLVL   16
#define TSIZE  (1u << 19)          // hashmap entries per level
#define TMASK  (TSIZE - 1u)

__device__ __forceinline__ float elu1(float v) {
    return v > 0.0f ? v : (__expf(v) - 1.0f);
}

__global__ __launch_bounds__(256) void
field_fwd_kernel(const float* __restrict__ x,      // (N,3)
                 const float* __restrict__ table,  // (L,T,2)
                 const float* __restrict__ W0,     // (35,32)
                 const float* __restrict__ b0,     // (32)
                 const float* __restrict__ W1,     // (32,32)
                 const float* __restrict__ b1,     // (32)
                 const float* __restrict__ Wout,   // (32,1)
                 const float* __restrict__ bout,   // (1)
                 float* __restrict__ out,          // (N,1)
                 int ntotal)
{
    // ---- LDS ----------------------------------------------------------------
    __shared__ alignas(16) _Float16 w0t[32][64];        // W0^T, K padded 35->64
    __shared__ alignas(16) _Float16 w1t[32][32];        // W1^T
    __shared__ alignas(16) _Float16 featA[8][16][64];   // per-wave A buffer (16xK64)
    __shared__ alignas(16) _Float16 h1s[8][16][32];     // per-wave hidden1 (16x32)

    const int tid  = threadIdx.x;
    const int wave = tid >> 5;
    const int lane = tid & 31;
    const int m    = lane & 15;     // row within 16-point tile / frag row
    const int hi   = lane >> 4;     // lane half-group

    // ---- stage weights (transposed, f16) -----------------------------------
    for (int i = tid; i < 32 * 64; i += 256) {
        int k = i & 63, n = i >> 6;
        w0t[n][k] = (k < 35) ? (_Float16)W0[k * 32 + n] : (_Float16)0.0f;
    }
    for (int i = tid; i < 32 * 32; i += 256) {
        int k = i & 31, n = i >> 5;
        w1t[n][k] = (_Float16)W1[k * 32 + n];
    }
    __syncthreads();

    // ---- per-lane point + coordinate normalization -------------------------
    const int base = blockIdx.x * 128 + wave * 16;
    int p = base + m;
    if (p > ntotal - 1) p = ntotal - 1;               // clamp tail (EXEC stays full)

    // Streaming reads: non-temporal so they never displace the L2-resident table.
    const float xr0 = __builtin_nontemporal_load(x + 3 * p + 0);
    const float xr1 = __builtin_nontemporal_load(x + 3 * p + 1);
    const float xr2 = __builtin_nontemporal_load(x + 3 * p + 2);
    const float xn0 = (xr0 + 1.0f) * 0.5f;            // SCALE = 1
    const float xn1 = (xr1 + 1.0f) * 0.5f;
    const float xn2 = (xr2 + 1.0f) * 0.5f;

    if (hi == 0) {                                    // K 0..2 = coords
        featA[wave][m][0] = (_Float16)xn0;
        featA[wave][m][1] = (_Float16)xn1;
        featA[wave][m][2] = (_Float16)xn2;
    } else {                                          // zero pad K 35..63
        for (int k = 35; k < 64; ++k) featA[wave][m][k] = (_Float16)0.0f;
    }

    // ---- hash-grid encode: this lane handles 8 levels ----------------------
    const float2* __restrict__ tbl2 = (const float2*)table;
    #pragma unroll
    for (int j = 0; j < 8; ++j) {
        const int level = hi * 8 + j;
        // B = exp(ln(512/16)/15) = 2^(1/3)  =>  s = 16*2^(level/3) - 1
        const float s = 16.0f * exp2f((float)level * (1.0f / 3.0f)) - 1.0f;

        const float px = xn0 * s, py = xn1 * s, pz = xn2 * s;
        const float fx = floorf(px), fy = floorf(py), fz = floorf(pz);
        const float wx = px - fx, wy = py - fy, wz = pz - fz;
        const unsigned ix = (unsigned)(int)fx;
        const unsigned iy = (unsigned)(int)fy;
        const unsigned iz = (unsigned)(int)fz;

        float f0 = 0.0f, f1 = 0.0f;
        #pragma unroll
        for (int c = 0; c < 8; ++c) {
            const unsigned cx = c & 1, cy = (c >> 1) & 1, cz = (c >> 2) & 1;
            const unsigned idx =
                ((ix + cx) * 1u ^
                 (iy + cy) * 2654435761u ^
                 (iz + cz) * 805459861u) & TMASK;
            const float cw = (cx ? wx : 1.0f - wx) *
                             (cy ? wy : 1.0f - wy) *
                             (cz ? wz : 1.0f - wz);
            // Regular-temporal gather: keep the table hot in L2.
            const float2 g = tbl2[(size_t)level * TSIZE + idx];  // global_load_b64
            f0 += cw * g.x;
            f1 += cw * g.y;
        }
        featA[wave][m][3 + 2 * level + 0] = (_Float16)f0;
        featA[wave][m][3 + 2 * level + 1] = (_Float16)f1;
    }
    __syncthreads();

    // ---- layer 0: (16x64) @ (64x32) via 4 WMMAs ----------------------------
    const unsigned* fa  = (const unsigned*)&featA[wave][0][0];  // stride 32 u32/row
    const unsigned* w0u = (const unsigned*)&w0t[0][0];          // stride 32 u32/row
    const unsigned* w1u = (const unsigned*)&w1t[0][0];          // stride 16 u32/row

    Frag16 a0, a1;                       // A fragments, K-blocks 0 and 32
    #pragma unroll
    for (int t = 0; t < 8; ++t) {
        const int off = hi * 8 + t;
        a0.u[t] = fa[m * 32 +      off];
        a1.u[t] = fa[m * 32 + 16 + off];
    }

    Frag16 bA0, bA1, bB0, bB1;           // B frags: [N-tile][K-block]
    #pragma unroll
    for (int t = 0; t < 8; ++t) {
        const int n0 = m, n1 = 16 + m;
        bA0.u[t] = w0u[n0 * 32 +      hi * 8 + t];
        bA1.u[t] = w0u[n0 * 32 + 16 + hi * 8 + t];
        bB0.u[t] = w0u[n1 * 32 +      hi * 8 + t];
        bB1.u[t] = w0u[n1 * 32 + 16 + hi * 8 + t];
    }

    const float b0lo = b0[m], b0hi16 = b0[16 + m];
    v8f acc0 = {b0lo, b0lo, b0lo, b0lo, b0lo, b0lo, b0lo, b0lo};
    v8f acc1 = {b0hi16, b0hi16, b0hi16, b0hi16, b0hi16, b0hi16, b0hi16, b0hi16};

    acc0 = __builtin_amdgcn_wmma_f32_16x16x32_f16(false, a0.h, false, bA0.h,
                                                  (short)0, acc0, false, false);
    acc0 = __builtin_amdgcn_wmma_f32_16x16x32_f16(false, a1.h, false, bA1.h,
                                                  (short)0, acc0, false, false);
    acc1 = __builtin_amdgcn_wmma_f32_16x16x32_f16(false, a0.h, false, bB0.h,
                                                  (short)0, acc1, false, false);
    acc1 = __builtin_amdgcn_wmma_f32_16x16x32_f16(false, a1.h, false, bB1.h,
                                                  (short)0, acc1, false, false);

    // ELU -> restage h1 (C/D layout: VGPR r holds rows r / r+8)
    #pragma unroll
    for (int r = 0; r < 8; ++r) {
        const int row = r + hi * 8;
        h1s[wave][row][m]      = (_Float16)elu1(acc0[r]);
        h1s[wave][row][16 + m] = (_Float16)elu1(acc1[r]);
    }
    __syncthreads();

    // ---- layer 1: (16x32) @ (32x32) via 2 WMMAs ----------------------------
    const unsigned* h1u = (const unsigned*)&h1s[wave][0][0];    // stride 16 u32/row
    Frag16 a2, bC0, bC1;
    #pragma unroll
    for (int t = 0; t < 8; ++t) {
        const int off = hi * 8 + t;
        a2.u[t]  = h1u[m * 16 + off];
        bC0.u[t] = w1u[m * 16 + hi * 8 + t];          // N-tile 0: n = m
        bC1.u[t] = w1u[(16 + m) * 16 + hi * 8 + t];   // N-tile 1: n = 16+m
    }

    const float b1lo = b1[m], b1hi16 = b1[16 + m];
    v8f h2a = {b1lo, b1lo, b1lo, b1lo, b1lo, b1lo, b1lo, b1lo};
    v8f h2b = {b1hi16, b1hi16, b1hi16, b1hi16, b1hi16, b1hi16, b1hi16, b1hi16};

    h2a = __builtin_amdgcn_wmma_f32_16x16x32_f16(false, a2.h, false, bC0.h,
                                                 (short)0, h2a, false, false);
    h2b = __builtin_amdgcn_wmma_f32_16x16x32_f16(false, a2.h, false, bC1.h,
                                                 (short)0, h2b, false, false);

    // ---- output layer: (32 -> 1) per-lane MAC + 16-lane xor reduction ------
    const float woLo = Wout[m];
    const float woHi = Wout[16 + m];
    const float bo   = bout[0];

    #pragma unroll
    for (int r = 0; r < 8; ++r) {
        float v = elu1(h2a[r]) * woLo + elu1(h2b[r]) * woHi;
        v += __shfl_xor(v, 1, 32);
        v += __shfl_xor(v, 2, 32);
        v += __shfl_xor(v, 4, 32);
        v += __shfl_xor(v, 8, 32);
        if (m == 0) {
            const int row = base + r + hi * 8;
            if (row < ntotal) {
                // Streaming write: non-temporal, don't pollute L2.
                __builtin_nontemporal_store(v + bo, out + row);
            }
        }
    }
}

extern "C" void kernel_launch(void* const* d_in, const int* in_sizes, int n_in,
                              void* d_out, int out_size, void* d_ws, size_t ws_size,
                              hipStream_t stream) {
    const float* x     = (const float*)d_in[0];
    const float* table = (const float*)d_in[1];
    const float* W0    = (const float*)d_in[2];
    const float* b0    = (const float*)d_in[3];
    const float* W1    = (const float*)d_in[4];
    const float* b1    = (const float*)d_in[5];
    const float* Wout  = (const float*)d_in[6];
    const float* bout  = (const float*)d_in[7];
    float* out = (float*)d_out;

    const int n = in_sizes[0] / 3;            // 2,000,000
    const int nblocks = (n + 127) / 128;      // 128 points per block (8 waves x 16)

    field_fwd_kernel<<<nblocks, 256, 0, stream>>>(
        x, table, W0, b0, W1, b1, Wout, bout, out, n);
}
